// SearchTrans_69861938037583
// MI455X (gfx1250) — matrix operand order repaired and checked by reading, not compiled, not run on hardware
//
#include <hip/hip_runtime.h>
#include <hip/hip_bf16.h>

// ---------------------------------------------------------------------------
// TTSR search (cosine-sim argmax over 4096x4096, K=2304, via bf16 WMMA) +
// transfer (gather + fold) for MI455X (gfx1250, wave32).
//
// Sizes: B=2, C3=256, H=W=64, L=4096, K=2304.
// Workspace layout (needs ~75.8 MB):
//   matA  (refsr patches, normalized bf16) : 2*4096*2304*2 = 37,748,736 B
//   matB  (lrsr  patches, normalized bf16) : 37,748,736 B
//   ss    (per-pixel channel sum-of-squares, 2 srcs * 2 batches * 4096 f32)
//   packed(u64 max-keys, 2*4096)
//   arg   (int argmax,   2*4096)
// ---------------------------------------------------------------------------

typedef unsigned short ushort_t;
typedef __attribute__((ext_vector_type(16))) __bf16 v16bf;
typedef __attribute__((ext_vector_type(8)))  float  v8f;

#define KDIM 2304
#define NPOS 4096

union Frag {
    uint4 q[2];
    v16bf v;
};

__device__ __forceinline__ unsigned short f2bf(float f) {
    unsigned u = __float_as_uint(f);
    unsigned r = u + 0x7FFFu + ((u >> 16) & 1u);   // round-to-nearest-even
    return (unsigned short)(r >> 16);
}

__device__ __forceinline__ unsigned mono_key(float f) {
    // order-preserving map float -> u32 (handles negatives)
    unsigned u = __float_as_uint(f);
    return (u & 0x80000000u) ? ~u : (u | 0x80000000u);
}

// --------------------------------------------------------------------------
// 1) per-pixel sum of squares over channels: ss[which][b][pos]
//    which=0 -> lrsr (B matrix), which=1 -> refsr (A matrix)
// --------------------------------------------------------------------------
__global__ __launch_bounds__(256)
void ss_kernel(const float* __restrict__ lrsr, const float* __restrict__ refsr,
               float* __restrict__ ss) {
    int idx   = blockIdx.x * 256 + threadIdx.x;    // 16384 threads total
    int pos   = idx & 4095;
    int b     = (idx >> 12) & 1;
    int which = idx >> 13;
    const float* src = (which ? refsr : lrsr) + (size_t)b * 256 * 4096 + pos;
    float s = 0.f;
    for (int c = 0; c < 256; ++c) {
        float v = src[(size_t)c * 4096];
        s += v * v;
    }
    ss[idx] = s;
}

// --------------------------------------------------------------------------
// 2) unfold 3x3 patch, normalize, pack bf16 row-major [pos][k], k = c*9 + j
// --------------------------------------------------------------------------
__global__ __launch_bounds__(256)
void pack_kernel(const float* __restrict__ lrsr, const float* __restrict__ refsr,
                 const float* __restrict__ ss,
                 ushort_t* __restrict__ matB, ushort_t* __restrict__ matA) {
    int blk   = blockIdx.x;            // 16384 blocks: [which][b][pos]
    int pos   = blk & 4095;
    int b     = (blk >> 12) & 1;
    int which = blk >> 13;
    int py = pos >> 6, px = pos & 63;

    const float* src = (which ? refsr : lrsr) + (size_t)b * 256 * 4096;
    const float* ssp = ss + ((size_t)which * 2 + b) * 4096;

    float ssum = 0.f;
#pragma unroll
    for (int j = 0; j < 9; ++j) {
        int sy = py + j / 3 - 1, sx = px + j % 3 - 1;
        if ((unsigned)sy < 64u && (unsigned)sx < 64u) ssum += ssp[sy * 64 + sx];
    }
    float inv = 1.f / fmaxf(sqrtf(ssum), 1e-12f);

    ushort_t* dst = (which ? matA : matB) + ((size_t)b * NPOS + pos) * KDIM;
    int t = threadIdx.x;
#pragma unroll
    for (int i = 0; i < 9; ++i) {
        int k = t + i * 256;
        int c = k / 9, j = k % 9;
        int sy = py + j / 3 - 1, sx = px + j % 3 - 1;
        float v = 0.f;
        if ((unsigned)sy < 64u && (unsigned)sx < 64u)
            v = src[((size_t)c << 12) + sy * 64 + sx];
        dst[k] = f2bf(v * inv);
    }
}

// --------------------------------------------------------------------------
// 3) zero-init packed keys
// --------------------------------------------------------------------------
__global__ __launch_bounds__(256)
void init_kernel(unsigned long long* __restrict__ packed) {
    int i = blockIdx.x * 256 + threadIdx.x;        // 8192
    packed[i] = 0ull;
}

// --------------------------------------------------------------------------
// 4) GEMM + streaming column max/argmax via bf16 WMMA.
//    grid (16 colBlocks, 8 rowBlocks, 2 batches), 256 threads = 8 waves.
//    Each wave: 2x2 tile (2 row fragments x 2 col fragments = 4 accumulators)
//    over a 512-row slab; K-loop of 72 steps issues 4 WMMAs per 8 b128 loads
//    (2.0 loads/WMMA). Running best kept in registers; merged across
//    row-blocks with packed u64 atomicMax (value-key | 4095-row for ties).
// --------------------------------------------------------------------------
__global__ __launch_bounds__(256)
void gemm_argmax_kernel(const ushort_t* __restrict__ matA,
                        const ushort_t* __restrict__ matB,
                        unsigned long long* __restrict__ packed) {
    int b       = blockIdx.z;
    int wave    = threadIdx.x >> 5;
    int lane    = threadIdx.x & 31;
    int laneRow = lane & 15;
    int half    = lane >> 4;             // 0: lanes 0-15, 1: lanes 16-31

    const ushort_t* A  = matA + (size_t)b * NPOS * KDIM;
    const ushort_t* Bm = matB + (size_t)b * NPOS * KDIM;

    int col0   = blockIdx.x * 256 + wave * 32;     // two col tiles: col0, col0+16
    int rowBeg = blockIdx.y * 512;

    // B-matrix fragment: lanes 0-15 K 0-7 & 16-23; lanes 16-31 K 8-15 & 24-31
    const ushort_t* bPtr0 = Bm + (size_t)(col0 + laneRow) * KDIM + half * 8;
    const ushort_t* bPtr1 = bPtr0 + (size_t)16 * KDIM;

    float bestV0 = -2.f, bestV1 = -2.f;
    int   bestR0 = 0,    bestR1 = 0;

    for (int rt = 0; rt < 512; rt += 32) {
        int row = rowBeg + rt;
        // A-matrix fragment: lanes 0-15 K 0-7 & 16-23; lanes 16-31 K 8-15 & 24-31
        const ushort_t* aPtr0 = A + (size_t)(row + laneRow) * KDIM + half * 8;
        const ushort_t* aPtr1 = aPtr0 + (size_t)16 * KDIM;
        v8f acc00 = {};   // rows row+0..15,  cols col0..col0+15
        v8f acc01 = {};   // rows row+0..15,  cols col0+16..col0+31
        v8f acc10 = {};   // rows row+16..31, cols col0..col0+15
        v8f acc11 = {};   // rows row+16..31, cols col0+16..col0+31
        for (int k = 0; k < KDIM; k += 32) {
            Frag a0, a1, b0, b1;
            a0.q[0] = *(const uint4*)(aPtr0 + k);
            a0.q[1] = *(const uint4*)(aPtr0 + k + 16);
            a1.q[0] = *(const uint4*)(aPtr1 + k);
            a1.q[1] = *(const uint4*)(aPtr1 + k + 16);
            b0.q[0] = *(const uint4*)(bPtr0 + k);
            b0.q[1] = *(const uint4*)(bPtr0 + k + 16);
            b1.q[0] = *(const uint4*)(bPtr1 + k);
            b1.q[1] = *(const uint4*)(bPtr1 + k + 16);
            acc00 = __builtin_amdgcn_wmma_f32_16x16x32_bf16(
                false, a0.v, false, b0.v, (short)0, acc00, false, false);
            acc01 = __builtin_amdgcn_wmma_f32_16x16x32_bf16(
                false, a0.v, false, b1.v, (short)0, acc01, false, false);
            acc10 = __builtin_amdgcn_wmma_f32_16x16x32_bf16(
                false, a1.v, false, b0.v, (short)0, acc10, false, false);
            acc11 = __builtin_amdgcn_wmma_f32_16x16x32_bf16(
                false, a1.v, false, b1.v, (short)0, acc11, false, false);
        }
        // C/D layout: lane holds rows (rbase + r), column col+laneRow.
        // Process lower row tile before upper one so strict '>' keeps the
        // smallest row index on ties (matches argmax semantics).
        int rbase0 = row + half * 8;
        int rbase1 = rbase0 + 16;
#pragma unroll
        for (int r = 0; r < 8; ++r) {
            float v00 = acc00[r];
            if (v00 > bestV0) { bestV0 = v00; bestR0 = rbase0 + r; }
            float v01 = acc01[r];
            if (v01 > bestV1) { bestV1 = v01; bestR1 = rbase0 + r; }
        }
#pragma unroll
        for (int r = 0; r < 8; ++r) {
            float v10 = acc10[r];
            if (v10 > bestV0) { bestV0 = v10; bestR0 = rbase1 + r; }
            float v11 = acc11[r];
            if (v11 > bestV1) { bestV1 = v11; bestR1 = rbase1 + r; }
        }
    }

    // combine lane l with lane l^16 (same column, other 8 rows); smaller row wins ties
    float oV0 = __shfl_xor(bestV0, 16);
    int   oR0 = __shfl_xor(bestR0, 16);
    if (oV0 > bestV0 || (oV0 == bestV0 && oR0 < bestR0)) { bestV0 = oV0; bestR0 = oR0; }
    float oV1 = __shfl_xor(bestV1, 16);
    int   oR1 = __shfl_xor(bestR1, 16);
    if (oV1 > bestV1 || (oV1 == bestV1 && oR1 < bestR1)) { bestV1 = oV1; bestR1 = oR1; }

    if (half == 0) {
        unsigned long long k0 =
            ((unsigned long long)mono_key(bestV0) << 32) | (unsigned)(4095 - bestR0);
        unsigned long long k1 =
            ((unsigned long long)mono_key(bestV1) << 32) | (unsigned)(4095 - bestR1);
        atomicMax(&packed[(size_t)b * NPOS + col0 + laneRow], k0);
        atomicMax(&packed[(size_t)b * NPOS + col0 + 16 + laneRow], k1);
    }
}

// --------------------------------------------------------------------------
// 5) decode packed keys -> S (star) and arg
// --------------------------------------------------------------------------
__global__ __launch_bounds__(256)
void finalize_kernel(const unsigned long long* __restrict__ packed,
                     float* __restrict__ S, int* __restrict__ arg) {
    int i = blockIdx.x * 256 + threadIdx.x;        // 8192
    unsigned long long k = packed[i];
    unsigned hi = (unsigned)(k >> 32);
    unsigned bits = (hi & 0x80000000u) ? (hi ^ 0x80000000u) : ~hi;
    S[i]   = __uint_as_float(bits);
    arg[i] = 4095 - (int)(unsigned)(k & 0xFFFFFFFFu);
}

// --------------------------------------------------------------------------
// 6) gather + fold. For kernel size 3s, pad s, stride s:
//    out(b,c,y,x) = (1/9) * sum over yo,xo in {floor(y/s)-1..+1} x {..} of
//                   ref(b,c, s*pyo + y - s*yo, s*pxo + x - s*xo)   (bounds-checked)
//    where p = arg[b][yo*64+xo], pyo=p>>6, pxo=p&63.
// --------------------------------------------------------------------------
template <int SS, int CC>
__global__ __launch_bounds__(256)
void fold_kernel(const float* __restrict__ ref, const int* __restrict__ arg,
                 float* __restrict__ out) {
    const int HO = 64 * SS;
    int idx = blockIdx.x * 256 + threadIdx.x;
    int x = idx % HO;
    int y = (idx / HO) % HO;
    int c = (idx / (HO * HO)) % CC;
    int b = idx / (HO * HO * CC);

    const int*   ab = arg + b * NPOS;
    const float* rb = ref + (size_t)(b * CC + c) * HO * HO;

    int qy = y / SS, qx = x / SS;
    float sum = 0.f;
#pragma unroll
    for (int dy = -1; dy <= 1; ++dy) {
        int yo = qy + dy;
        if ((unsigned)yo >= 64u) continue;
#pragma unroll
        for (int dx = -1; dx <= 1; ++dx) {
            int xo = qx + dx;
            if ((unsigned)xo >= 64u) continue;
            int p  = ab[yo * 64 + xo];
            int sy = SS * (p >> 6)  + y - SS * yo;
            int sx = SS * (p & 63) + x - SS * xo;
            if ((unsigned)sy < (unsigned)HO && (unsigned)sx < (unsigned)HO)
                sum += rb[sy * HO + sx];
        }
    }
    out[idx] = sum * (1.f / 9.f);
}

// --------------------------------------------------------------------------
// launch
// --------------------------------------------------------------------------
extern "C" void kernel_launch(void* const* d_in, const int* in_sizes, int n_in,
                              void* d_out, int out_size, void* d_ws, size_t ws_size,
                              hipStream_t stream) {
    (void)in_sizes; (void)n_in; (void)out_size; (void)ws_size;

    const float* lrsr  = (const float*)d_in[0];   // (2,256,64,64)
    const float* refsr = (const float*)d_in[1];   // (2,256,64,64)
    const float* ref1  = (const float*)d_in[2];   // (2,64,256,256)
    const float* ref2  = (const float*)d_in[3];   // (2,128,128,128)
    const float* ref3  = (const float*)d_in[4];   // (2,256,64,64)

    float* out   = (float*)d_out;
    float* outS  = out;                            // 2*1*64*64      = 8192
    float* outT3 = outS  + 8192;                   // 2*256*64*64    = 2097152
    float* outT2 = outT3 + 2097152;                // 2*128*128*128  = 4194304
    float* outT1 = outT2 + 4194304;                // 2*64*256*256   = 8388608

    char* ws = (char*)d_ws;
    const size_t MAT_BYTES = (size_t)2 * NPOS * KDIM * 2;   // 37,748,736
    ushort_t* matA = (ushort_t*)ws;                         // refsr patches
    ushort_t* matB = (ushort_t*)(ws + MAT_BYTES);           // lrsr patches
    float* ssbuf   = (float*)(ws + 2 * MAT_BYTES);          // 65,536 B
    unsigned long long* packed =
        (unsigned long long*)(ws + 2 * MAT_BYTES + 65536);  // 65,536 B
    int* arg = (int*)(ws + 2 * MAT_BYTES + 131072);         // 32,768 B

    ss_kernel<<<64, 256, 0, stream>>>(lrsr, refsr, ssbuf);
    pack_kernel<<<16384, 256, 0, stream>>>(lrsr, refsr, ssbuf, matB, matA);
    init_kernel<<<32, 256, 0, stream>>>(packed);
    gemm_argmax_kernel<<<dim3(16, 8, 2), 256, 0, stream>>>(matA, matB, packed);
    finalize_kernel<<<32, 256, 0, stream>>>(packed, outS, arg);
    fold_kernel<1, 256><<<8192,  256, 0, stream>>>(ref3, arg, outT3);
    fold_kernel<2, 128><<<16384, 256, 0, stream>>>(ref2, arg, outT2);
    fold_kernel<4, 64> <<<32768, 256, 0, stream>>>(ref1, arg, outT1);
}
